// MoDEEncoderBlock_12902081757309
// MI455X (gfx1250) — compile-verified, matrix operand order checked by reading, not run on hardware
//
#include <hip/hip_runtime.h>

typedef __attribute__((ext_vector_type(16))) _Float16 v16h;
typedef __attribute__((ext_vector_type(8)))  float    v8f;

// Problem constants (match reference setup_inputs)
#define NB   2
#define CO   64
#define CI1  32
#define CI2  64
#define TDIM 10
#define DSP  64
#define PD   68                                   // padded spatial extent (+2 halo each side)
#define SSPAT ((size_t)(DSP)*(DSP)*(DSP))         // 262144
#define SPAD  ((size_t)(PD)*(PD)*(PD))            // 314432
#define SOUT  ((size_t)32*32*32)                  // 32768

// ---------------------------------------------------------------------------
// 1) Gate: g[n][e][co] = softmax_e( (t @ gw^T + gb)[n, e*Co+co] )
// ---------------------------------------------------------------------------
__global__ void gate_kernel(const float* __restrict__ t, const float* __restrict__ gw,
                            const float* __restrict__ gb, float* __restrict__ g) {
  int idx = blockIdx.x * blockDim.x + threadIdx.x;   // over NB*CO
  if (idx >= NB * CO) return;
  int n = idx / CO, co = idx % CO;
  float logits[5];
  #pragma unroll
  for (int e = 0; e < 5; ++e) {
    int row = e * CO + co;
    float acc = gb[row];
    #pragma unroll
    for (int k = 0; k < TDIM; ++k) acc += t[n * TDIM + k] * gw[row * TDIM + k];
    logits[e] = acc;
  }
  float m = logits[0];
  #pragma unroll
  for (int e = 1; e < 5; ++e) m = fmaxf(m, logits[e]);
  float ex[5], s = 0.f;
  #pragma unroll
  for (int e = 0; e < 5; ++e) { ex[e] = __expf(logits[e] - m); s += ex[e]; }
  float inv = 1.f / s;
  #pragma unroll
  for (int e = 0; e < 5; ++e) g[(n * 5 + e) * CO + co] = ex[e] * inv;
}

// ---------------------------------------------------------------------------
// 2) Synthesize per-sample 5x5x5 kernels -> f16, layout [n][kpos][co][ci]
// ---------------------------------------------------------------------------
__global__ void synth_kernel(const float* __restrict__ c5, const float* __restrict__ c3,
                             const float* __restrict__ c1, const float* __restrict__ a3,
                             const float* __restrict__ a5, const float* __restrict__ g,
                             _Float16* __restrict__ wsyn, int Ci) {
  int idx = blockIdx.x * blockDim.x + threadIdx.x;   // over NB*CO*Ci
  if (idx >= NB * CO * Ci) return;
  int ci = idx % Ci;
  int co = (idx / Ci) % CO;
  int n  = idx / (Ci * CO);
  float g0 = g[(n * 5 + 0) * CO + co];
  float g1 = g[(n * 5 + 1) * CO + co];
  float g2 = g[(n * 5 + 2) * CO + co];
  float g3 = g[(n * 5 + 3) * CO + co];
  float g4 = g[(n * 5 + 4) * CO + co];
  float v1  = c1[co * Ci + ci];
  float va3 = a3[co * Ci + ci] * (1.f / 27.f);
  float va5 = a5[co * Ci + ci] * (1.f / 125.f);
  const float* p5 = c5 + (size_t)(co * Ci + ci) * 125;
  const float* p3 = c3 + (size_t)(co * Ci + ci) * 27;
  for (int kd = 0; kd < 5; ++kd)
    for (int kh = 0; kh < 5; ++kh)
      for (int kw = 0; kw < 5; ++kw) {
        float w = g0 * p5[kd * 25 + kh * 5 + kw] + g4 * va5;
        if (kd >= 1 && kd <= 3 && kh >= 1 && kh <= 3 && kw >= 1 && kw <= 3)
          w += g1 * p3[(kd - 1) * 9 + (kh - 1) * 3 + (kw - 1)] + g3 * va3;
        if (kd == 2 && kh == 2 && kw == 2) w += g2 * v1;
        int kpos = kd * 25 + kh * 5 + kw;
        wsyn[((size_t)(n * 125 + kpos) * CO + co) * Ci + ci] = (_Float16)w;
      }
}

// ---------------------------------------------------------------------------
// 3) x (f32, NCDHW) -> padded channels-last f16: xpad[n][pz][py][pw][ci]
//    (halo of 2 zero-filled on every spatial side; removes all bounds checks
//     and f32->f16 conversion from the WMMA hot loop)
// ---------------------------------------------------------------------------
__global__ void xpose_pad_kernel(const float* __restrict__ x, _Float16* __restrict__ xpad,
                                 int Ci) {
  size_t idx = (size_t)blockIdx.x * blockDim.x + threadIdx.x;   // over NB*SPAD
  if (idx >= (size_t)NB * SPAD) return;
  int n = (int)(idx / SPAD);
  size_t r = idx % SPAD;
  int pz = (int)(r / (PD * PD));
  int py = (int)((r / PD) % PD);
  int pw = (int)(r % PD);
  int z = pz - 2, y = py - 2, w = pw - 2;
  _Float16* dst = xpad + idx * Ci;
  bool in = ((unsigned)z < DSP) && ((unsigned)y < DSP) && ((unsigned)w < DSP);
  size_t sp = (size_t)z * DSP * DSP + (size_t)y * DSP + w;
  for (int ci = 0; ci < Ci; ++ci) {
    float v = in ? x[((size_t)n * Ci + ci) * SSPAT + sp] : 0.f;
    dst[ci] = (_Float16)v;
  }
}

// ---------------------------------------------------------------------------
// 4) Implicit-GEMM mode conv (WMMA f32 <- f16), SAME pad=2, 5x5x5.
//    Block = 128 threads (4 waves). Wave owns a 16-row co tile and FOUR
//    16-column tiles (whole W row of 64): one A fragment feeds 4 WMMAs.
//    B fragments are single contiguous 32B loads from the padded CL buffer.
// ---------------------------------------------------------------------------
template <int CI>
__global__ void __launch_bounds__(128)
conv_mode_cl(const _Float16* __restrict__ xpad, const _Float16* __restrict__ wsyn,
             float* __restrict__ y) {
  int h  = blockIdx.x;
  int dn = blockIdx.y;
  int d  = dn % DSP;
  int n  = dn / DSP;

  int lane  = threadIdx.x & 31;
  int wave  = threadIdx.x >> 5;
  int m0    = wave * 16;
  int mrow  = m0 + (lane & 15);
  int khalf = (lane >> 4) * 16;     // K sub-block this lane carries
  int col   = lane & 15;            // column within a 16-wide tile

  v8f acc[4] = {};

  const _Float16* xb = xpad + (size_t)n * SPAD * CI + (size_t)khalf;
  const _Float16* wn = wsyn + (size_t)n * 125 * CO * CI;

  for (int kd = 0; kd < 5; ++kd) {
    for (int kh = 0; kh < 5; ++kh) {
      // padded input row: pz = d+kd, py = h+kh (pad 2 cancels the -2 offset)
      const _Float16* brow =
          xb + (((size_t)(d + kd) * PD + (h + kh)) * PD + col) * CI;
      #pragma unroll
      for (int kw = 0; kw < 5; ++kw) {
        int kpos = kd * 25 + kh * 5 + kw;
        const _Float16* wb = wn + ((size_t)kpos * CO + mrow) * CI + khalf;
        __builtin_prefetch(wb + (size_t)CO * CI, 0, 1);   // next kpos weights
        #pragma unroll
        for (int kc = 0; kc < CI; kc += 32) {
          v16h a = *(const v16h*)(wb + kc);
          #pragma unroll
          for (int ct = 0; ct < 4; ++ct) {
            v16h b = *(const v16h*)(brow + (size_t)(ct * 16 + kw) * CI + kc);
            acc[ct] = __builtin_amdgcn_wmma_f32_16x16x32_f16(
                false, a, false, b, (short)0, acc[ct], false, false);
          }
        }
      }
    }
  }

  int mbase = m0 + ((lane >> 4) << 3);
  float* yb = y + (size_t)n * CO * SSPAT + (size_t)d * DSP * DSP + (size_t)h * DSP;
  #pragma unroll
  for (int ct = 0; ct < 4; ++ct)
    #pragma unroll
    for (int r = 0; r < 8; ++r)
      yb[(size_t)(mbase + r) * SSPAT + ct * 16 + col] = acc[ct][r];
}

// ---------------------------------------------------------------------------
// 5) Per-(n,c) mean / rstd over S elements
// ---------------------------------------------------------------------------
__global__ void stats_kernel(const float* __restrict__ y, float* __restrict__ stats,
                             size_t S) {
  __shared__ float ssum[256], ssq[256];
  const float* p = y + (size_t)blockIdx.x * S;
  float s = 0.f, q = 0.f;
  for (size_t i = threadIdx.x; i < S; i += blockDim.x) {
    float v = p[i];
    s += v; q += v * v;
  }
  ssum[threadIdx.x] = s; ssq[threadIdx.x] = q;
  __syncthreads();
  for (int off = 128; off > 0; off >>= 1) {
    if ((int)threadIdx.x < off) {
      ssum[threadIdx.x] += ssum[threadIdx.x + off];
      ssq[threadIdx.x]  += ssq[threadIdx.x + off];
    }
    __syncthreads();
  }
  if (threadIdx.x == 0) {
    float mu  = ssum[0] / (float)S;
    float var = ssq[0] / (float)S - mu * mu;
    stats[blockIdx.x * 2 + 0] = mu;
    stats[blockIdx.x * 2 + 1] = rsqrtf(var + 1e-5f);
  }
}

__device__ __forceinline__ float mishf(float v) {
  float sp = (v > 20.f) ? v : log1pf(__expf(v));   // softplus = logaddexp(v,0)
  return v * tanhf(sp);
}

// ---------------------------------------------------------------------------
// 6a) norm + mish -> padded channels-last f16 (feeds the next mode conv)
// ---------------------------------------------------------------------------
__global__ void norm_mish_pad_kernel(const float* __restrict__ y, const float* __restrict__ stats,
                                     const float* __restrict__ gamma, const float* __restrict__ beta,
                                     _Float16* __restrict__ hpad) {
  size_t i = (size_t)blockIdx.x * blockDim.x + threadIdx.x;    // [n][z][y][w][c], c fastest
  if (i >= (size_t)NB * CO * SSPAT) return;
  int c = (int)(i % CO);
  size_t sp = (i / CO) % SSPAT;
  int n = (int)(i / ((size_t)CO * SSPAT));
  size_t nc = (size_t)n * CO + c;
  float mu = stats[nc * 2 + 0], rs = stats[nc * 2 + 1];
  float v = (y[nc * SSPAT + sp] - mu) * rs * gamma[c] + beta[c];
  int z = (int)(sp / (DSP * DSP)), yy = (int)((sp / DSP) % DSP), w = (int)(sp % DSP);
  size_t pidx = (((size_t)n * PD + (z + 2)) * PD + (yy + 2)) * PD + (w + 2);
  hpad[pidx * CO + c] = (_Float16)mishf(v);
}

__global__ void zero_f16_kernel(_Float16* __restrict__ p, size_t n) {
  size_t i = (size_t)blockIdx.x * blockDim.x + threadIdx.x;
  if (i < n) p[i] = (_Float16)0.f;
}

// ---------------------------------------------------------------------------
// 6b) norm + mish -> f32 in place (x_skip output) AND unpadded CL f16 copy
//     (feeds the down-conv)
// ---------------------------------------------------------------------------
__global__ void norm_mish_dual_kernel(float* __restrict__ y, const float* __restrict__ stats,
                                      const float* __restrict__ gamma, const float* __restrict__ beta,
                                      _Float16* __restrict__ xt) {
  size_t i = (size_t)blockIdx.x * blockDim.x + threadIdx.x;    // [n][c][sp], sp fastest
  if (i >= (size_t)NB * CO * SSPAT) return;
  size_t nc = i / SSPAT;
  size_t sp = i % SSPAT;
  int c = (int)(nc % CO);
  int n = (int)(nc / CO);
  float mu = stats[nc * 2 + 0], rs = stats[nc * 2 + 1];
  float v = (y[i] - mu) * rs * gamma[c] + beta[c];
  float m = mishf(v);
  y[i] = m;
  xt[((size_t)n * SSPAT + sp) * CO + c] = (_Float16)m;
}

// ---------------------------------------------------------------------------
// 6c) plain norm + mish in place (final y)
// ---------------------------------------------------------------------------
__global__ void norm_mish_kernel(float* __restrict__ y, const float* __restrict__ stats,
                                 const float* __restrict__ gamma, const float* __restrict__ beta,
                                 size_t S, int C, size_t total) {
  size_t i = (size_t)blockIdx.x * blockDim.x + threadIdx.x;
  if (i >= total) return;
  size_t nc = i / S;
  int c = (int)(nc % C);
  float mu = stats[nc * 2 + 0], rs = stats[nc * 2 + 1];
  float v = (y[i] - mu) * rs * gamma[c] + beta[c];
  y[i] = mishf(v);
}

// ---------------------------------------------------------------------------
// 7) Down-conv weight: (Co,Co,2,2,2) f32 -> f16 layout [kpos][co][ci]
// ---------------------------------------------------------------------------
__global__ void synth_down(const float* __restrict__ dw, _Float16* __restrict__ wd) {
  int idx = blockIdx.x * blockDim.x + threadIdx.x;    // over CO*CO
  if (idx >= CO * CO) return;
  int ci = idx % CO, co = idx / CO;
  #pragma unroll
  for (int k = 0; k < 8; ++k)
    wd[((size_t)k * CO + co) * CO + ci] = (_Float16)dw[(size_t)(co * CO + ci) * 8 + k];
}

// ---------------------------------------------------------------------------
// 8) Stride-2 2x2x2 VALID conv via WMMA, channels-last f16 input.
//    Block = 4 waves; wave owns a co tile and TWO 16-col tiles (full 32-w row).
// ---------------------------------------------------------------------------
__global__ void __launch_bounds__(128)
conv_down_cl(const _Float16* __restrict__ xt, const _Float16* __restrict__ wd,
             float* __restrict__ y) {
  int h  = blockIdx.x;          // 0..31
  int dn = blockIdx.y;
  int d  = dn % 32;
  int n  = dn / 32;
  int lane  = threadIdx.x & 31;
  int wave  = threadIdx.x >> 5;
  int m0    = wave * 16;
  int mrow  = m0 + (lane & 15);
  int khalf = (lane >> 4) * 16;
  int col   = lane & 15;
  v8f acc[2] = {};
  const _Float16* xb = xt + (size_t)n * SSPAT * CO + khalf;

  #pragma unroll
  for (int kd = 0; kd < 2; ++kd)
    #pragma unroll
    for (int kh = 0; kh < 2; ++kh) {
      const _Float16* brow =
          xb + (((size_t)(2 * d + kd) * DSP + (2 * h + kh)) * DSP) * CO;
      #pragma unroll
      for (int kw = 0; kw < 2; ++kw) {
        int kpos = kd * 4 + kh * 2 + kw;
        const _Float16* wb = wd + ((size_t)kpos * CO + mrow) * CO + khalf;
        #pragma unroll
        for (int kc = 0; kc < CO; kc += 32) {
          v16h a = *(const v16h*)(wb + kc);
          #pragma unroll
          for (int ct = 0; ct < 2; ++ct) {
            int wg = 2 * (ct * 16 + col) + kw;
            v16h b = *(const v16h*)(brow + (size_t)wg * CO + kc);
            acc[ct] = __builtin_amdgcn_wmma_f32_16x16x32_f16(
                false, a, false, b, (short)0, acc[ct], false, false);
          }
        }
      }
    }

  int mbase = m0 + ((lane >> 4) << 3);
  float* yb = y + (size_t)n * CO * SOUT + (size_t)d * 32 * 32 + (size_t)h * 32;
  #pragma unroll
  for (int ct = 0; ct < 2; ++ct)
    #pragma unroll
    for (int r = 0; r < 8; ++r)
      yb[(size_t)(mbase + r) * SOUT + ct * 16 + col] = acc[ct][r];
}

// ---------------------------------------------------------------------------
extern "C" void kernel_launch(void* const* d_in, const int* in_sizes, int n_in,
                              void* d_out, int out_size, void* d_ws, size_t ws_size,
                              hipStream_t stream) {
  const float* x        = (const float*)d_in[0];
  const float* t        = (const float*)d_in[1];
  const float* l1_c5    = (const float*)d_in[2];
  const float* l1_c3    = (const float*)d_in[3];
  const float* l1_c1    = (const float*)d_in[4];
  const float* l1_a3    = (const float*)d_in[5];
  const float* l1_a5    = (const float*)d_in[6];
  const float* l1_gw    = (const float*)d_in[7];
  const float* l1_gb    = (const float*)d_in[8];
  const float* l1_gamma = (const float*)d_in[9];
  const float* l1_beta  = (const float*)d_in[10];
  const float* l2_c5    = (const float*)d_in[11];
  const float* l2_c3    = (const float*)d_in[12];
  const float* l2_c1    = (const float*)d_in[13];
  const float* l2_a3    = (const float*)d_in[14];
  const float* l2_a5    = (const float*)d_in[15];
  const float* l2_gw    = (const float*)d_in[16];
  const float* l2_gb    = (const float*)d_in[17];
  const float* l2_gamma = (const float*)d_in[18];
  const float* l2_beta  = (const float*)d_in[19];
  const float* down_w   = (const float*)d_in[20];
  const float* dn_gamma = (const float*)d_in[21];
  const float* dn_beta  = (const float*)d_in[22];
  float* out = (float*)d_out;

  // d_out: [ y_down : NB*CO*32^3 ][ x_skip : NB*CO*64^3 ]
  const size_t y_elems  = (size_t)NB * CO * SOUT;      // 4,194,304
  float* out_y    = out;
  float* out_skip = out + y_elems;

  // workspace carve-up
  char* ws = (char*)d_ws;
  size_t off = 0;
  auto carve = [&](size_t bytes) -> char* {
    char* p = ws + off;
    off += (bytes + 255) & ~(size_t)255;
    return p;
  };
  float*    g1    = (float*)   carve((size_t)NB * 5 * CO * sizeof(float));
  float*    g2    = (float*)   carve((size_t)NB * 5 * CO * sizeof(float));
  float*    stats = (float*)   carve((size_t)NB * CO * 2 * sizeof(float));
  _Float16* wsyn1 = (_Float16*)carve((size_t)NB * 125 * CO * CI1 * sizeof(_Float16));
  _Float16* wsyn2 = (_Float16*)carve((size_t)NB * 125 * CO * CI2 * sizeof(_Float16));
  _Float16* wdn   = (_Float16*)carve((size_t)8 * CO * CO * sizeof(_Float16));
  _Float16* xpad1 = (_Float16*)carve((size_t)NB * SPAD * CI1 * sizeof(_Float16));
  _Float16* hpad2 = (_Float16*)carve((size_t)NB * SPAD * CI2 * sizeof(_Float16));
  _Float16* xskpt = (_Float16*)carve((size_t)NB * SSPAT * CO * sizeof(_Float16));
  float*    hraw  = (float*)   carve((size_t)NB * CO * SSPAT * sizeof(float));
  (void)ws_size; (void)in_sizes; (void)n_in; (void)out_size;

  const size_t tot1 = (size_t)NB * CO * SSPAT;   // 33,554,432
  const size_t totd = y_elems;

  // ---- layer 1 ----
  gate_kernel<<<1, 128, 0, stream>>>(t, l1_gw, l1_gb, g1);
  synth_kernel<<<(NB * CO * CI1 + 255) / 256, 256, 0, stream>>>(
      l1_c5, l1_c3, l1_c1, l1_a3, l1_a5, g1, wsyn1, CI1);
  xpose_pad_kernel<<<(unsigned)((NB * SPAD + 255) / 256), 256, 0, stream>>>(x, xpad1, CI1);
  conv_mode_cl<CI1><<<dim3(DSP, NB * DSP), 128, 0, stream>>>(xpad1, wsyn1, hraw);
  stats_kernel<<<NB * CO, 256, 0, stream>>>(hraw, stats, SSPAT);
  {
    size_t npad = (size_t)NB * SPAD * CI2;
    zero_f16_kernel<<<(unsigned)((npad + 255) / 256), 256, 0, stream>>>(hpad2, npad);
  }
  norm_mish_pad_kernel<<<(unsigned)((tot1 + 255) / 256), 256, 0, stream>>>(
      hraw, stats, l1_gamma, l1_beta, hpad2);

  // ---- layer 2 (raw output straight into x_skip slot of d_out) ----
  gate_kernel<<<1, 128, 0, stream>>>(t, l2_gw, l2_gb, g2);
  synth_kernel<<<(NB * CO * CI2 + 255) / 256, 256, 0, stream>>>(
      l2_c5, l2_c3, l2_c1, l2_a3, l2_a5, g2, wsyn2, CI2);
  conv_mode_cl<CI2><<<dim3(DSP, NB * DSP), 128, 0, stream>>>(hpad2, wsyn2, out_skip);
  stats_kernel<<<NB * CO, 256, 0, stream>>>(out_skip, stats, SSPAT);
  norm_mish_dual_kernel<<<(unsigned)((tot1 + 255) / 256), 256, 0, stream>>>(
      out_skip, stats, l2_gamma, l2_beta, xskpt);

  // ---- downsample ----
  synth_down<<<(CO * CO + 255) / 256, 256, 0, stream>>>(down_w, wdn);
  conv_down_cl<<<dim3(32, NB * 32), 128, 0, stream>>>(xskpt, wdn, out_y);
  stats_kernel<<<NB * CO, 256, 0, stream>>>(out_y, stats, SOUT);
  norm_mish_kernel<<<(unsigned)((totd + 255) / 256), 256, 0, stream>>>(
      out_y, stats, dn_gamma, dn_beta, SOUT, CO, totd);
}
